// MultiHeadAttention_79628693667982
// MI455X (gfx1250) — matrix operand order repaired and checked by reading, not compiled, not run on hardware
//
#include <hip/hip_runtime.h>

// ---------------------------------------------------------------------------
// MI455X / gfx1250 fused multi-head attention, bf16 WMMA (f32 accumulate)
// Async global->LDS staging (ASYNCcnt) + double-buffered tiles.
// ---------------------------------------------------------------------------

typedef __attribute__((ext_vector_type(16))) __bf16 v16bf;
typedef __attribute__((ext_vector_type(8)))  __bf16 v8bf;
typedef __attribute__((ext_vector_type(4)))  __bf16 v4bf;
typedef __attribute__((ext_vector_type(8)))  float  v8f;

#define B_   2
#define N_   2048
#define C_   1024
#define H_   16
#define HD_  64
#define O3_  3072            // 3*H*HD
#define M_   (B_ * N_)       // 4096

#if __has_builtin(__builtin_amdgcn_global_load_async_to_lds_b128) && \
    __has_builtin(__builtin_amdgcn_s_wait_asynccnt)
#define USE_ASYNC_LDS 1
#else
#define USE_ASYNC_LDS 0
#endif

#if USE_ASYNC_LDS
typedef int v4i_t __attribute__((vector_size(16)));
typedef __attribute__((address_space(1))) v4i_t as1_v4i;   // HIP __device__ (global)
typedef __attribute__((address_space(3))) v4i_t as3_v4i;   // HIP __shared__ (LDS)
#endif

// 16-byte global -> LDS copy (async on CDNA5: no VGPR bounce, ASYNCcnt).
__device__ __forceinline__ void cp_async16(const __bf16* g, __bf16* l) {
#if USE_ASYNC_LDS
  __builtin_amdgcn_global_load_async_to_lds_b128(
      (as1_v4i*)g, (as3_v4i*)l, 0, 0);
#else
  *(uint4*)l = *(const uint4*)g;
#endif
}
__device__ __forceinline__ void cp_wait() {
#if USE_ASYNC_LDS
  __builtin_amdgcn_s_wait_asynccnt(0);
#endif
}

// ---------------- WMMA helpers (CDNA5 ISA 7.12.2 layouts) ------------------

__device__ __forceinline__ v8f vzero8() {
  v8f z;
#pragma unroll
  for (int j = 0; j < 8; ++j) z[j] = 0.0f;
  return z;
}

__device__ __forceinline__ v8f wmma_bf16(v16bf a, v16bf b, v8f c) {
  return __builtin_amdgcn_wmma_f32_16x16x32_bf16(
      false, a, false, b, (short)0, c, false, false);
}

// A fragment, 16x32 bf16: lane L -> row = L%16, g = L/16.
// elements 0..7 = K (g*8 + 0..7), elements 8..15 = K (16 + g*8 + 0..7).
__device__ __forceinline__ v16bf frag_a_ld(const __bf16* base, int stride, int lane) {
  const int r = lane & 15, g = lane >> 4;
  const __bf16* p = base + (size_t)r * stride + g * 8;
  v8bf lo = *(const v8bf*)(p);
  v8bf hi = *(const v8bf*)(p + 16);
  return __builtin_shufflevector(lo, hi, 0, 1, 2, 3, 4, 5, 6, 7,
                                 8, 9, 10, 11, 12, 13, 14, 15);
}

// B fragment, 32x16 bf16, source stored as W[col][k] row-major:
// lane L -> col = L%16, g = L/16: elements 0..7 = K (g*8 + 0..7), 8..15 = K (16 + g*8 + 0..7).
__device__ __forceinline__ v16bf frag_b_ld(const __bf16* base, int stride, int lane) {
  const __bf16* p = base + (size_t)(lane & 15) * stride + (lane >> 4) * 8;
  v8bf lo = *(const v8bf*)(p);
  v8bf hi = *(const v8bf*)(p + 16);
  return __builtin_shufflevector(lo, hi, 0, 1, 2, 3, 4, 5, 6, 7,
                                 8, 9, 10, 11, 12, 13, 14, 15);
}

// ---------------- small kernels --------------------------------------------

__global__ void k_cvt_bf16(const float* __restrict__ in, __bf16* __restrict__ out,
                           int n4) {
  int i = blockIdx.x * 256 + threadIdx.x;
  if (i < n4) {
    float4 v = ((const float4*)in)[i];
    v4bf o = { (__bf16)v.x, (__bf16)v.y, (__bf16)v.z, (__bf16)v.w };
    ((v4bf*)out)[i] = o;
  }
}

__global__ void k_mean(const float* __restrict__ x, float* __restrict__ mean_x) {
  const int b = blockIdx.x >> 2;
  const int c = (blockIdx.x & 3) * 256 + threadIdx.x;
  const float* p = x + (size_t)b * N_ * C_ + c;
  float s = 0.0f;
  for (int n = 0; n < N_; ++n) s += p[(size_t)n * C_];
  mean_x[b * C_ + c] = s * (1.0f / (float)N_);
}

// gates = sigmoid(gate_w @ mean + gate_b); coef = gates * softmax(head_weights);
// qk_scale[h] = scale / softplus(dynamic_temp[h])     (BASE_TEMP == 1)
__global__ void k_prep(const float* __restrict__ mean_x,
                       const float* __restrict__ gate_w,
                       const float* __restrict__ gate_b,
                       const float* __restrict__ dynamic_temp,
                       const float* __restrict__ head_weights,
                       const float* __restrict__ scale,
                       float* __restrict__ qk_scale,
                       float* __restrict__ coef,
                       float* __restrict__ hw) {
  const int t = threadIdx.x;
  if (t < H_) {
    float dt = dynamic_temp[t];
    float sp = (dt > 20.0f) ? dt : log1pf(expf(dt));      // softplus
    qk_scale[t] = scale[0] / sp;
    float mx = -INFINITY;
    for (int i = 0; i < H_; ++i) mx = fmaxf(mx, head_weights[i]);
    float sum = 0.0f;
    for (int i = 0; i < H_; ++i) sum += expf(head_weights[i] - mx);
    hw[t] = expf(head_weights[t] - mx) / sum;
  }
  __syncthreads();
  const int p = t >> 3, l8 = t & 7;   // 32 (b,h) pairs, 8 lanes each
  const int b = p >> 4, h = p & 15;
  float s = 0.0f;
  for (int c = l8; c < C_; c += 8)
    s += mean_x[b * C_ + c] * gate_w[h * C_ + c];
  s += __shfl_xor(s, 1, 32);
  s += __shfl_xor(s, 2, 32);
  s += __shfl_xor(s, 4, 32);
  if (l8 == 0) {
    float g = 1.0f / (1.0f + expf(-(s + gate_b[h])));
    coef[b * H_ + h] = g * hw[h];
  }
}

// ---------------- shared GEMM tile stage -----------------------------------
// 128x32 bf16 A and B tiles; 4 x 16B async chunks per thread.

__device__ __forceinline__ void stage_tile(const __bf16* __restrict__ gA,
                                           const __bf16* __restrict__ gB,
                                           __bf16 (*lA)[32], __bf16 (*lB)[32],
                                           int t, int m0, int n0, int k0) {
#pragma unroll
  for (int i = 0; i < 2; ++i) {
    int c = t + i * 256;
    int row = c >> 2, part = c & 3;
    cp_async16(&gA[(size_t)(m0 + row) * C_ + k0 + part * 8], &lA[row][part * 8]);
    cp_async16(&gB[(size_t)(n0 + row) * C_ + k0 + part * 8], &lB[row][part * 8]);
  }
}

// ---------------- QKV projection GEMM: [4096,1024] x [3072,1024]^T ---------
// Epilogue scatters q/k as [B,H,N,HD] and v transposed as [B,H,HD,N] (bf16).

__global__ __launch_bounds__(256) void k_gemm_qkv(const __bf16* __restrict__ xb,
                                                  const __bf16* __restrict__ wb,
                                                  __bf16* __restrict__ qh,
                                                  __bf16* __restrict__ kh,
                                                  __bf16* __restrict__ vT) {
  __shared__ __align__(16) __bf16 As[2][128][32];
  __shared__ __align__(16) __bf16 Bs[2][128][32];
  const int t = threadIdx.x;
  const int lane = t & 31, w = t >> 5;
  const int wm = w & 1, wn = w >> 1;     // 2 x 4 wave grid
  const int m0 = blockIdx.x * 128;
  const int n0 = blockIdx.y * 128;

  v8f acc[4][2];
#pragma unroll
  for (int mi = 0; mi < 4; ++mi)
#pragma unroll
    for (int ni = 0; ni < 2; ++ni) acc[mi][ni] = vzero8();

  stage_tile(xb, wb, As[0], Bs[0], t, m0, n0, 0);
  const int nk = C_ / 32;
#pragma unroll          // compile-time trip count: fixed acc registers, no PHIs
  for (int ki = 0; ki < nk; ++ki) {
    const int buf = ki & 1;
    cp_wait();
    __syncthreads();
    if (ki + 1 < nk)
      stage_tile(xb, wb, As[buf ^ 1], Bs[buf ^ 1], t, m0, n0, (ki + 1) * 32);

    v16bf bf0 = frag_b_ld(&Bs[buf][wn * 32 + 0][0], 32, lane);
    v16bf bf1 = frag_b_ld(&Bs[buf][wn * 32 + 16][0], 32, lane);
    v16bf af0 = frag_a_ld(&As[buf][wm * 64 + 0][0], 32, lane);
    v16bf af1 = frag_a_ld(&As[buf][wm * 64 + 16][0], 32, lane);
    v16bf af2 = frag_a_ld(&As[buf][wm * 64 + 32][0], 32, lane);
    v16bf af3 = frag_a_ld(&As[buf][wm * 64 + 48][0], 32, lane);
    acc[0][0] = wmma_bf16(af0, bf0, acc[0][0]);
    acc[0][1] = wmma_bf16(af0, bf1, acc[0][1]);
    acc[1][0] = wmma_bf16(af1, bf0, acc[1][0]);
    acc[1][1] = wmma_bf16(af1, bf1, acc[1][1]);
    acc[2][0] = wmma_bf16(af2, bf0, acc[2][0]);
    acc[2][1] = wmma_bf16(af2, bf1, acc[2][1]);
    acc[3][0] = wmma_bf16(af3, bf0, acc[3][0]);
    acc[3][1] = wmma_bf16(af3, bf1, acc[3][1]);
  }

#pragma unroll
  for (int mi = 0; mi < 4; ++mi)
#pragma unroll
    for (int ni = 0; ni < 2; ++ni)
#pragma unroll
      for (int j = 0; j < 8; ++j) {
        int gm = m0 + wm * 64 + mi * 16 + j + 8 * (lane >> 4);
        int gc = n0 + wn * 32 + ni * 16 + (lane & 15);
        int b = gm >> 11, n = gm & (N_ - 1);
        int sel = gc >> 10, rem = gc & 1023;
        int h = rem >> 6, d = rem & 63;
        __bf16 v = (__bf16)acc[mi][ni][j];
        size_t rowm = ((size_t)(b * H_ + h) * N_ + n) * HD_ + d;
        if (sel == 0)      qh[rowm] = v;
        else if (sel == 1) kh[rowm] = v;
        else               vT[((size_t)(b * H_ + h) * HD_ + d) * N_ + n] = v;
      }
}

// ---------------- Flash attention per (b, h, 128 query rows) ---------------

__global__ __launch_bounds__(256) void k_attn(const __bf16* __restrict__ qh,
                                              const __bf16* __restrict__ kh,
                                              const __bf16* __restrict__ vT,
                                              const float* __restrict__ qk_scale,
                                              const float* __restrict__ coef,
                                              __bf16* __restrict__ ao) {
  __shared__ __align__(16) __bf16 Ks[2][32][64];    // [key][d]
  __shared__ __align__(16) __bf16 Vs[2][64][32];    // [d][key]  (from vT)
  __shared__ __align__(16) __bf16 Ps[8][16][32];    // per-wave P relayout
  const int t = threadIdx.x, lane = t & 31, w = t >> 5;
  const int b = blockIdx.z, h = blockIdx.y, q0 = blockIdx.x * 128;
  const int bh = b * H_ + h;
  const int qrow = q0 + w * 16;

  const __bf16* qbase = qh + ((size_t)bh * N_ + qrow) * HD_;
  const __bf16* kbase = kh + (size_t)bh * N_ * HD_;
  const __bf16* vbase = vT + (size_t)bh * HD_ * N_;

  const int krow = t >> 3, kpart = t & 7;           // K tile: 32 x 64
  const int vrow = t >> 2, vpart = t & 3;           // V^T tile: 64 x 32

  v16bf qf0 = frag_a_ld(qbase, HD_, lane);          // d = 0..31
  v16bf qf1 = frag_a_ld(qbase + 32, HD_, lane);     // d = 32..63
  const float sc = qk_scale[h];

  float mrow[8], lrow[8];
  v8f o[4];
#pragma unroll
  for (int j = 0; j < 8; ++j) { mrow[j] = -INFINITY; lrow[j] = 0.0f; }
#pragma unroll
  for (int f = 0; f < 4; ++f) o[f] = vzero8();

  // prologue stage of tile 0
  cp_async16(&kbase[(size_t)krow * HD_ + kpart * 8], &Ks[0][krow][kpart * 8]);
  cp_async16(&vbase[(size_t)vrow * N_ + vpart * 8], &Vs[0][vrow][vpart * 8]);

  // one 32-key step; `buf` is always a literal constant at the call sites
  auto attn_step = [&](int buf, int kt) {
    cp_wait();
    __syncthreads();
    if (kt + 32 < N_) {
      cp_async16(&kbase[(size_t)(kt + 32 + krow) * HD_ + kpart * 8],
                 &Ks[buf ^ 1][krow][kpart * 8]);
      cp_async16(&vbase[(size_t)vrow * N_ + kt + 32 + vpart * 8],
                 &Vs[buf ^ 1][vrow][vpart * 8]);
    }

    v16bf kb00 = frag_b_ld(&Ks[buf][0][0],  64, lane);
    v16bf kb01 = frag_b_ld(&Ks[buf][0][32], 64, lane);
    v16bf kb10 = frag_b_ld(&Ks[buf][16][0], 64, lane);
    v16bf kb11 = frag_b_ld(&Ks[buf][16][32], 64, lane);
    v8f s0 = vzero8(), s1 = vzero8();               // keys kt+0..15 / +16..31
    s0 = wmma_bf16(qf0, kb00, s0);
    s0 = wmma_bf16(qf1, kb01, s0);
    s1 = wmma_bf16(qf0, kb10, s1);
    s1 = wmma_bf16(qf1, kb11, s1);

    // fetch V fragments while doing softmax VALU work
    v16bf vb0 = frag_b_ld(&Vs[buf][0][0],  32, lane);
    v16bf vb1 = frag_b_ld(&Vs[buf][16][0], 32, lane);
    v16bf vb2 = frag_b_ld(&Vs[buf][32][0], 32, lane);
    v16bf vb3 = frag_b_ld(&Vs[buf][48][0], 32, lane);

    float corr[8];
#pragma unroll
    for (int j = 0; j < 8; ++j) {
      float a0 = s0[j] * sc, a1 = s1[j] * sc;
      float mx = fmaxf(a0, a1);                     // row max over 16 lanes
      mx = fmaxf(mx, __shfl_xor(mx, 1, 32));
      mx = fmaxf(mx, __shfl_xor(mx, 2, 32));
      mx = fmaxf(mx, __shfl_xor(mx, 4, 32));
      mx = fmaxf(mx, __shfl_xor(mx, 8, 32));
      float newm = fmaxf(mrow[j], mx);
      float p0 = __expf(a0 - newm);
      float p1 = __expf(a1 - newm);
      float cr = __expf(mrow[j] - newm);
      float ps = p0 + p1;                           // row sum over 16 lanes
      ps += __shfl_xor(ps, 1, 32);
      ps += __shfl_xor(ps, 2, 32);
      ps += __shfl_xor(ps, 4, 32);
      ps += __shfl_xor(ps, 8, 32);
      lrow[j] = lrow[j] * cr + ps;
      mrow[j] = newm;
      corr[j] = cr;
      int r = j + 8 * (lane >> 4);                  // C-layout -> LDS [row][key]
      Ps[w][r][(lane & 15)]      = (__bf16)p0;
      Ps[w][r][16 + (lane & 15)] = (__bf16)p1;
    }
#pragma unroll
    for (int f = 0; f < 4; ++f)
#pragma unroll
      for (int j = 0; j < 8; ++j) o[f][j] *= corr[j];

    v16bf pf = frag_a_ld(&Ps[w][0][0], 32, lane);   // A-layout reload of P
    o[0] = wmma_bf16(pf, vb0, o[0]);
    o[1] = wmma_bf16(pf, vb1, o[1]);
    o[2] = wmma_bf16(pf, vb2, o[2]);
    o[3] = wmma_bf16(pf, vb3, o[3]);
  };

  for (int kt = 0; kt < N_; kt += 64) {             // two steps, literal bufs
    attn_step(0, kt);
    attn_step(1, kt + 32);
  }

  const float cf = coef[bh];
  float inv[8];
#pragma unroll
  for (int j = 0; j < 8; ++j) inv[j] = cf / lrow[j];
#pragma unroll
  for (int f = 0; f < 4; ++f)
#pragma unroll
    for (int j = 0; j < 8; ++j) {
      int n = qrow + j + 8 * (lane >> 4);
      int d = f * 16 + (lane & 15);
      ao[((size_t)(b * N_) + n) * (H_ * HD_) + h * HD_ + d] =
          (__bf16)(o[f][j] * inv[j]);
    }
}

// ---------------- output projection GEMM: [4096,1024] x [1024,1024]^T + b --

__global__ __launch_bounds__(256) void k_gemm_out(const __bf16* __restrict__ ab,
                                                  const __bf16* __restrict__ wb,
                                                  const float* __restrict__ bias,
                                                  float* __restrict__ out) {
  __shared__ __align__(16) __bf16 As[2][128][32];
  __shared__ __align__(16) __bf16 Bs[2][128][32];
  const int t = threadIdx.x;
  const int lane = t & 31, w = t >> 5;
  const int wm = w & 1, wn = w >> 1;
  const int m0 = blockIdx.x * 128;
  const int n0 = blockIdx.y * 128;

  v8f acc[4][2];
#pragma unroll
  for (int mi = 0; mi < 4; ++mi)
#pragma unroll
    for (int ni = 0; ni < 2; ++ni) acc[mi][ni] = vzero8();

  stage_tile(ab, wb, As[0], Bs[0], t, m0, n0, 0);
  const int nk = C_ / 32;
#pragma unroll          // compile-time trip count: fixed acc registers, no PHIs
  for (int ki = 0; ki < nk; ++ki) {
    const int buf = ki & 1;
    cp_wait();
    __syncthreads();
    if (ki + 1 < nk)
      stage_tile(ab, wb, As[buf ^ 1], Bs[buf ^ 1], t, m0, n0, (ki + 1) * 32);

    v16bf bf0 = frag_b_ld(&Bs[buf][wn * 32 + 0][0], 32, lane);
    v16bf bf1 = frag_b_ld(&Bs[buf][wn * 32 + 16][0], 32, lane);
    v16bf af0 = frag_a_ld(&As[buf][wm * 64 + 0][0], 32, lane);
    v16bf af1 = frag_a_ld(&As[buf][wm * 64 + 16][0], 32, lane);
    v16bf af2 = frag_a_ld(&As[buf][wm * 64 + 32][0], 32, lane);
    v16bf af3 = frag_a_ld(&As[buf][wm * 64 + 48][0], 32, lane);
    acc[0][0] = wmma_bf16(af0, bf0, acc[0][0]);
    acc[0][1] = wmma_bf16(af0, bf1, acc[0][1]);
    acc[1][0] = wmma_bf16(af1, bf0, acc[1][0]);
    acc[1][1] = wmma_bf16(af1, bf1, acc[1][1]);
    acc[2][0] = wmma_bf16(af2, bf0, acc[2][0]);
    acc[2][1] = wmma_bf16(af2, bf1, acc[2][1]);
    acc[3][0] = wmma_bf16(af3, bf0, acc[3][0]);
    acc[3][1] = wmma_bf16(af3, bf1, acc[3][1]);
  }

#pragma unroll
  for (int mi = 0; mi < 4; ++mi)
#pragma unroll
    for (int ni = 0; ni < 2; ++ni)
#pragma unroll
      for (int j = 0; j < 8; ++j) {
        int gm = m0 + wm * 64 + mi * 16 + j + 8 * (lane >> 4);
        int gc = n0 + wn * 32 + ni * 16 + (lane & 15);
        out[(size_t)gm * C_ + gc] = acc[mi][ni][j] + bias[gc];
      }
}

// ---------------------------------------------------------------------------

extern "C" void kernel_launch(void* const* d_in, const int* in_sizes, int n_in,
                              void* d_out, int out_size, void* d_ws, size_t ws_size,
                              hipStream_t stream) {
  (void)in_sizes; (void)n_in; (void)out_size; (void)ws_size;
  const float* x            = (const float*)d_in[0];
  const float* scale        = (const float*)d_in[1];
  const float* dynamic_temp = (const float*)d_in[2];
  const float* gate_w       = (const float*)d_in[3];
  const float* gate_b       = (const float*)d_in[4];
  const float* qkv_w        = (const float*)d_in[5];
  const float* out_w        = (const float*)d_in[6];
  const float* out_b        = (const float*)d_in[7];
  const float* head_weights = (const float*)d_in[8];
  float* out = (float*)d_out;

  char* ws = (char*)d_ws;
  __bf16* xb     = (__bf16*)(ws);                        //  8 MB  x bf16
  __bf16* qkvwb  = (__bf16*)(ws + ( 8ull << 20));        //  6 MB  qkv_w bf16
  __bf16* outwb  = (__bf16*)(ws + (14ull << 20));        //  2 MB  out_w bf16
  __bf16* qhb    = (__bf16*)(ws + (16ull << 20));        //  8 MB  q [B,H,N,HD]
  __bf16* khb    = (__bf16*)(ws + (20ull << 20));        //  8 MB  k [B,H,N,HD]
  __bf16* vTb    = (__bf16*)(ws + (24ull << 20));        //  8 MB  v [B,H,HD,N]
  __bf16* aob    = (__bf16*)(ws + (28ull << 20));        //  8 MB  attn out bf16
  float*  mean_x = (float*)(ws + (36ull << 20));         //  8 KB
  float*  qks    = mean_x + B_ * C_;
  float*  coefp  = qks + H_;
  float*  hwp    = coefp + B_ * H_;

  // 1) bf16 conversions
  k_cvt_bf16<<<(M_ * C_ / 4 + 255) / 256, 256, 0, stream>>>(x, xb, M_ * C_ / 4);
  k_cvt_bf16<<<(O3_ * C_ / 4 + 255) / 256, 256, 0, stream>>>(qkv_w, qkvwb, O3_ * C_ / 4);
  k_cvt_bf16<<<(C_ * C_ / 4 + 255) / 256, 256, 0, stream>>>(out_w, outwb, C_ * C_ / 4);

  // 2) gates / temps / head-weight softmax
  k_mean<<<B_ * (C_ / 256), 256, 0, stream>>>(x, mean_x);
  k_prep<<<1, 256, 0, stream>>>(mean_x, gate_w, gate_b, dynamic_temp,
                                head_weights, scale, qks, coefp, hwp);

  // 3) QKV projection (WMMA bf16, async double-buffered LDS)
  k_gemm_qkv<<<dim3(M_ / 128, O3_ / 128), 256, 0, stream>>>(xb, qkvwb, qhb, khb, vTb);

  // 4) flash attention (WMMA bf16, async double-buffered K/V tiles)
  k_attn<<<dim3(N_ / 128, H_, B_), 256, 0, stream>>>(qhb, khb, vTb, qks, coefp, aob);

  // 5) output projection + bias (WMMA bf16, f32 out)
  k_gemm_out<<<dim3(M_ / 128, C_ / 128), 256, 0, stream>>>(aob, outwb, out_b, out);
}
